// OCRModel_22600117912226
// MI455X (gfx1250) — matrix operand order repaired and checked by reading, not compiled, not run on hardware
//
#include <hip/hip_runtime.h>
#include <hip/hip_bf16.h>
#include <stdint.h>

typedef _Float16 half_t;
typedef __attribute__((ext_vector_type(16))) _Float16 v16h;
typedef __attribute__((ext_vector_type(8)))  _Float16 v8h;
typedef __attribute__((ext_vector_type(8)))  float    v8f;

static __device__ __forceinline__ float sigmoidf_(float x) {
  return 1.0f / (1.0f + __expf(-x));
}

// ---------------------------------------------------------------------------
// WMMA fragment layouts per CDNA5 ISA 7.12.2 (wave32):
//   A (16x32 f16): lane L -> row M=L&15; group g=L>>4 holds K = g*8 + {0..7}
//                  in v0..3 and K = g*8 + {16..23} in v4..7.
//   B (32x16 f16): lane L -> col N=L&15; group g holds K = g*8 + {0..7}, g*8 + {16..23}.
//   C/D (16x16 f32): vgpr r, lane L -> M = r + 8*(L>>4), N = L&15.
//
// B matrices are pre-packed so lane L's 16 halves for (n-tile nt, k-chunk kc)
// are contiguous: Bp[(((nt*KC)+kc)*32 + L)*16 + j]   (KC = K/32)
// ---------------------------------------------------------------------------

// ---------------------------------------------------------------------------
// Dense GEMM: C[MxN] = A[MxK](f16,row-major,lda) * Bpacked  (+C)(+bias)(+bias2)
// Requirements (all call sites satisfy): M%16==0, N%256==0, K%32==0,
// A rows 16B-aligned.
// Block = 128 threads (4 waves). Block tile: 16(M) x 256(N); each wave owns
// 4 consecutive 16-col tiles. The 16x32 A tile per k-chunk is staged into LDS
// with gfx1250 async global->LDS loads (double buffered, ASYNCcnt), then each
// lane reads its fragment back with ds_load_b128.
// ---------------------------------------------------------------------------
__global__ void __launch_bounds__(128)
gemm_wmma(const half_t* __restrict__ A, int lda,
          const half_t* __restrict__ Bp,
          float* __restrict__ C, int ldc,
          const float* __restrict__ bias,
          const float* __restrict__ bias2,
          int K, int accumulate)
{
  __shared__ half_t smem[2][16 * 32];

  const int tid  = threadIdx.x;
  const int lane = tid & 31;
  const int wave = tid >> 5;
  const int row  = lane & 15;
  const int grp  = lane >> 4;
  const int KC   = K >> 5;
  const int m0   = blockIdx.x * 16;
  const int nt0  = blockIdx.y * 16 + wave * 4;   // first 16-col tile of wave

  const half_t* Abase = A + (size_t)m0 * lda;

  // stage k-chunk kc of the block's 16x32 A tile into smem[buf] (async)
  auto issue = [&](int kc, int buf) {
    if (tid < 64) {                       // waves 0,1: 64 chunks of 16B
      int r = tid >> 2, part = tid & 3;
      const half_t* g = Abase + (size_t)r * lda + kc * 32 + part * 8;
      unsigned laddr = (unsigned)(uintptr_t)&smem[buf][r * 32 + part * 8];
      asm volatile("global_load_async_to_lds_b128 %0, %1, off"
                   :: "v"(laddr), "v"(g) : "memory");
    }
  };

  issue(0, 0);

  v8f acc[4] = {{}, {}, {}, {}};
  const size_t tile_stride = (size_t)KC * 512;   // halves between n-tiles

  for (int kc = 0; kc < KC; ++kc) {
    const int buf = kc & 1;
    asm volatile("s_wait_asynccnt 0x0" ::: "memory");
    __syncthreads();
    if (kc + 1 < KC) issue(kc + 1, buf ^ 1);

    // A fragment from LDS (two ds_load_b128)
    const v8h* ap = (const v8h*)&smem[buf][row * 32 + grp * 8];
    v8h alo = ap[0];
    v8h ahi = ap[2];     // +16 halves
    v16h a;
#pragma unroll
    for (int j = 0; j < 8; ++j) { a[j] = alo[j]; a[j + 8] = ahi[j]; }

    // B fragments: 32 contiguous bytes per lane per tile
    const half_t* bbase = Bp + ((size_t)nt0 * KC + kc) * 512 + lane * 16;
#pragma unroll
    for (int tt = 0; tt < 4; ++tt) {
      v16h b = *(const v16h*)(bbase + (size_t)tt * tile_stride);
      acc[tt] = __builtin_amdgcn_wmma_f32_16x16x32_f16(false, a, false, b,
                                                       (short)0, acc[tt],
                                                       false, false);
    }
  }

  // epilogue
#pragma unroll
  for (int tt = 0; tt < 4; ++tt) {
    int n = (nt0 + tt) * 16 + row;
#pragma unroll
    for (int r = 0; r < 8; ++r) {
      int m = m0 + grp * 8 + r;
      float v = acc[tt][r];
      if (accumulate) v += C[(size_t)m * ldc + n];
      if (bias)  v += bias[n];
      if (bias2) v += bias2[n];
      C[(size_t)m * ldc + n] = v;
    }
  }
}

// ---------------------------------------------------------------------------
// Implicit-GEMM conv 3x3 SAME + bias + ReLU via WMMA. One wave per block;
// each wave: 16 output positions x 64 output channels (4 n-tiles, A reuse).
// A gathered branchlessly from NCHW f32 input (clamped address + select).
// Weights pre-packed in fragment order over Kpad=roundup(Cin*9,32).
// Requirements: (Bn*Hout*Wout)%16==0, Cout%64==0.
// ---------------------------------------------------------------------------
__global__ void __launch_bounds__(32)
conv_relu_wmma(const float* __restrict__ in,
               const half_t* __restrict__ Wp,
               const float* __restrict__ bias,
               float* __restrict__ out,
               half_t* __restrict__ encx,
               int Bn, int Cin, int Hin, int Win,
               int Cout, int Hout, int Wout,
               int stride, int Kpad)
{
  const int lane = threadIdx.x;
  const int row  = lane & 15;
  const int grp  = lane >> 4;
  const int m0   = blockIdx.x * 16;
  const int nt0  = blockIdx.y * 4;
  const int K    = Cin * 9;
  const int KC   = Kpad >> 5;
  int tph = (Hout - 1) * stride + 3 - Hin; if (tph < 0) tph = 0;
  int tpw = (Wout - 1) * stride + 3 - Win; if (tpw < 0) tpw = 0;
  const int ph = tph / 2, pw = tpw / 2;

  const int m  = m0 + row;
  const int ox = m % Wout;
  const int oy = (m / Wout) % Hout;
  const int bb = m / (Wout * Hout);
  const float* inb = in + (size_t)bb * Cin * Hin * Win;

  v8f acc[4] = {{}, {}, {}, {}};
  const size_t tile_stride = (size_t)KC * 512;

  for (int kc = 0; kc < KC; ++kc) {
    const int k0 = kc * 32;
    v16h a;
#pragma unroll
    for (int j = 0; j < 16; ++j) {
      int k = k0 + grp * 8 + (j < 8 ? j : j + 8);
      bool kok = k < K;
      int kcl = kok ? k : 0;
      int cin = kcl / 9, rem = kcl % 9, kh = rem / 3, kw = rem % 3;
      int iy = oy * stride + kh - ph;
      int ix = ox * stride + kw - pw;
      bool ok = kok && (iy >= 0) && (iy < Hin) && (ix >= 0) && (ix < Win);
      int iyc = iy < 0 ? 0 : (iy >= Hin ? Hin - 1 : iy);
      int ixc = ix < 0 ? 0 : (ix >= Win ? Win - 1 : ix);
      float v = inb[((size_t)cin * Hin + iyc) * Win + ixc];  // unconditional
      a[j] = ok ? (half_t)v : (half_t)0.0f;                  // v_cndmask
    }
    const half_t* bbase = Wp + ((size_t)nt0 * KC + kc) * 512 + lane * 16;
    __builtin_prefetch(bbase + 512, 0, 1);
#pragma unroll
    for (int tt = 0; tt < 4; ++tt) {
      v16h b = *(const v16h*)(bbase + (size_t)tt * tile_stride);
      acc[tt] = __builtin_amdgcn_wmma_f32_16x16x32_f16(false, a, false, b,
                                                       (short)0, acc[tt],
                                                       false, false);
    }
  }

#pragma unroll
  for (int tt = 0; tt < 4; ++tt) {
    int co = (nt0 + tt) * 16 + row;
#pragma unroll
    for (int r = 0; r < 8; ++r) {
      int mo = m0 + grp * 8 + r;
      int ox2 = mo % Wout;
      int oy2 = (mo / Wout) % Hout;
      int b2  = mo / (Wout * Hout);
      float v = acc[tt][r] + bias[co];
      v = fmaxf(v, 0.0f);
      out[(((size_t)b2 * Cout + co) * Hout + oy2) * Wout + ox2] = v;
      if (encx) {
        size_t nrow = (size_t)b2 * Hout + oy2;
        encx[((size_t)ox2 * ((size_t)Bn * Hout) + nrow) * Cout + co] =
            (half_t)v;
      }
    }
  }
}

// ---- weight packing into WMMA-fragment order ------------------------------
// dense: W[N x K] f32 (row-major) -> Bp fragment order (B = W^T, KxN)
__global__ void pack_dense_frag(const float* __restrict__ W,
                                half_t* __restrict__ out, int N, int K)
{
  int idx = blockIdx.x * blockDim.x + threadIdx.x;
  if (idx >= N * K) return;
  int n = idx / K, k = idx % K;
  int KC = K >> 5;
  int nt = n >> 4, nl = n & 15;
  int kc = k >> 5, kk = k & 31, g = (kk >> 3) & 1, j = (kk & 7) | ((kk >> 4) << 3);
  size_t dst = (((size_t)nt * KC + kc) * 32 + (g * 16 + nl)) * 16 + j;
  out[dst] = (half_t)W[idx];
}

// conv: W[Cout,Cin,3,3] f32 -> fragment order over Kpad (zero-fill k>=Cin*9)
__global__ void pack_conv_frag(const float* __restrict__ W,
                               half_t* __restrict__ out,
                               int Cout, int Cin, int Kpad)
{
  int idx = blockIdx.x * blockDim.x + threadIdx.x;
  if (idx >= Cout * Kpad) return;
  int co = idx / Kpad, k = idx % Kpad;
  int K = Cin * 9;
  float v = (k < K) ? W[(size_t)co * K + k] : 0.0f;
  int KC = Kpad >> 5;
  int nt = co >> 4, nl = co & 15;
  int kc = k >> 5, kk = k & 31, g = (kk >> 3) & 1, j = (kk & 7) | ((kk >> 4) << 3);
  size_t dst = (((size_t)nt * KC + kc) * 32 + (g * 16 + nl)) * 16 + j;
  out[dst] = (half_t)v;
}

// ---- zero fill ------------------------------------------------------------
__global__ void zero_f32(float* __restrict__ p, int n)
{
  int i = blockIdx.x * blockDim.x + threadIdx.x;
  if (i < n) p[i] = 0.0f;
}
__global__ void zero_f16(half_t* __restrict__ p, int n)
{
  int i = blockIdx.x * blockDim.x + threadIdx.x;
  if (i < n) p[i] = (half_t)0.0f;
}

// ---- LSTM cell (pytorch gate order i,f,g,o) -------------------------------
__global__ void lstm_cell(const float* __restrict__ gates,
                          float* __restrict__ h, float* __restrict__ c,
                          half_t* __restrict__ h16,
                          int N, int H,
                          float* __restrict__ enc_f32,
                          half_t* __restrict__ enc_f16,
                          int t, int Hp, int Wp, int useenc)
{
  int idx = blockIdx.x * blockDim.x + threadIdx.x;
  if (idx >= N * H) return;
  int n = idx / H, j = idx % H;
  const float* g = gates + (size_t)n * 4 * H;
  float ig = sigmoidf_(g[j]);
  float fg = sigmoidf_(g[H + j]);
  float gg = tanhf(g[2 * H + j]);
  float og = sigmoidf_(g[3 * H + j]);
  float cc = fg * c[idx] + ig * gg;
  float hh = og * tanhf(cc);
  c[idx] = cc;
  h[idx] = hh;
  h16[idx] = (half_t)hh;
  if (useenc) {
    int b = n / Hp, hp = n % Hp;
    size_t l = (size_t)hp * Wp + t;
    size_t o = ((size_t)b * ((size_t)Hp * Wp) + l) * H + j;
    enc_f32[o] = hh;
    enc_f16[o] = (half_t)hh;
  }
}

// ---- attention ------------------------------------------------------------
__global__ void attn_scores(const float* __restrict__ enc_proj,
                            const float* __restrict__ query,
                            const float* __restrict__ vvec,
                            float* __restrict__ scores, int L, int H)
{
  int l = blockIdx.x, b = blockIdx.y, tid = threadIdx.x;
  const float* ep = enc_proj + ((size_t)b * L + l) * H;
  const float* q  = query + (size_t)b * H;
  float s = 0.0f;
  for (int hh = tid; hh < H; hh += 64) s += tanhf(ep[hh] + q[hh]) * vvec[hh];
  __shared__ float red[64];
  red[tid] = s; __syncthreads();
  for (int off = 32; off > 0; off >>= 1) {
    if (tid < off) red[tid] += red[tid + off];
    __syncthreads();
  }
  if (tid == 0) scores[(size_t)b * L + l] = red[0];
}

__global__ void softmax_rows(const float* __restrict__ scores,
                             float* __restrict__ attn, int L)
{
  int b = blockIdx.x, tid = threadIdx.x;
  const float* s = scores + (size_t)b * L;
  __shared__ float red[256];
  float m = -1e30f;
  for (int l = tid; l < L; l += 256) m = fmaxf(m, s[l]);
  red[tid] = m; __syncthreads();
  for (int off = 128; off > 0; off >>= 1) {
    if (tid < off) red[tid] = fmaxf(red[tid], red[tid + off]);
    __syncthreads();
  }
  float mx = red[0]; __syncthreads();
  float sum = 0.0f;
  for (int l = tid; l < L; l += 256) sum += __expf(s[l] - mx);
  red[tid] = sum; __syncthreads();
  for (int off = 128; off > 0; off >>= 1) {
    if (tid < off) red[tid] += red[tid + off];
    __syncthreads();
  }
  float tot = red[0];
  for (int l = tid; l < L; l += 256)
    attn[(size_t)b * L + l] = __expf(s[l] - mx) / tot;
}

__global__ void context_kernel(const float* __restrict__ attn,
                               const float* __restrict__ enc,
                               float* __restrict__ ctx, int L, int H)
{
  int hcol = blockIdx.x * blockDim.x + threadIdx.x;
  int b = blockIdx.y;
  if (hcol >= H) return;
  const float* a = attn + (size_t)b * L;
  const float* e = enc + (size_t)b * L * H + hcol;
  float s = 0.0f;
  for (int l = 0; l < L; ++l) s += a[l] * e[(size_t)l * H];
  ctx[(size_t)b * H + hcol] = s;
}

__global__ void build_x0(const int* __restrict__ targets, int t, int TGT,
                         const float* __restrict__ emb, int EMBd,
                         const float* __restrict__ ctx,
                         half_t* __restrict__ x0, int Bn, int Hd)
{
  int idx = blockIdx.x * blockDim.x + threadIdx.x;
  int cols = EMBd + Hd;
  if (idx >= Bn * cols) return;
  int b = idx / cols, col = idx % cols;
  float v;
  if (col < EMBd) {
    int tok = targets[(size_t)b * TGT + t];
    v = emb[(size_t)tok * EMBd + col];
  } else {
    v = ctx[(size_t)b * Hd + (col - EMBd)];
  }
  x0[idx] = (half_t)v;
}

__global__ void nll_kernel(const float* __restrict__ logits,
                           const int* __restrict__ targets, int t, int TGT,
                           float* __restrict__ loss, int V)
{
  int b = blockIdx.x, tid = threadIdx.x;
  const float* lg = logits + (size_t)b * V;
  __shared__ float red[256];
  float m = -1e30f;
  for (int j = tid; j < V; j += 256) m = fmaxf(m, lg[j]);
  red[tid] = m; __syncthreads();
  for (int off = 128; off > 0; off >>= 1) {
    if (tid < off) red[tid] = fmaxf(red[tid], red[tid + off]);
    __syncthreads();
  }
  float mx = red[0]; __syncthreads();
  float s = 0.0f;
  for (int j = tid; j < V; j += 256) s += __expf(lg[j] - mx);
  red[tid] = s; __syncthreads();
  for (int off = 128; off > 0; off >>= 1) {
    if (tid < off) red[tid] += red[tid + off];
    __syncthreads();
  }
  if (tid == 0) {
    int tok = targets[(size_t)b * TGT + t];
    if (tok != 1) {
      float lp = lg[tok] - mx - __logf(red[0]);
      atomicAdd(loss, -lp);
    }
  }
}

__global__ void finalize_kernel(const float* __restrict__ loss,
                                float* __restrict__ out, float invB)
{
  out[0] = loss[0] * invB;
}

// ===========================================================================
extern "C" void kernel_launch(void* const* d_in, const int* in_sizes, int n_in,
                              void* d_out, int out_size, void* d_ws, size_t ws_size,
                              hipStream_t stream)
{
  (void)in_sizes; (void)n_in; (void)out_size; (void)ws_size;

  const int Bn = 32, HIMG = 128, WIMG = 512;
  const int VOCAB = 512, EMBd = 256, HID = 512, TGT = 128;
  const int Hp = 16, Wp = 64, L = Hp * Wp;      // 1024
  const int Nenc = Bn * Hp;                     // 512

  const float* images  = (const float*)d_in[0];
  const int*   targets = (const int*)d_in[1];
  const float* cw[4] = {(const float*)d_in[2], (const float*)d_in[4],
                        (const float*)d_in[6], (const float*)d_in[8]};
  const float* cb[4] = {(const float*)d_in[3], (const float*)d_in[5],
                        (const float*)d_in[7], (const float*)d_in[9]};
  const float* enc_Wih = (const float*)d_in[10];
  const float* enc_Whh = (const float*)d_in[11];
  const float* enc_bih = (const float*)d_in[12];
  const float* enc_bhh = (const float*)d_in[13];
  const float* att_Wenc = (const float*)d_in[14];
  const float* att_Wdec = (const float*)d_in[15];
  const float* att_v    = (const float*)d_in[16];
  const float* emb      = (const float*)d_in[17];
  const float* dec_Wih0 = (const float*)d_in[18];
  const float* dec_Whh0 = (const float*)d_in[19];
  const float* dec_bih0 = (const float*)d_in[20];
  const float* dec_bhh0 = (const float*)d_in[21];
  const float* dec_Wih1 = (const float*)d_in[22];
  const float* dec_Whh1 = (const float*)d_in[23];
  const float* dec_bih1 = (const float*)d_in[24];
  const float* dec_bhh1 = (const float*)d_in[25];
  const float* out_W    = (const float*)d_in[26];
  const float* out_b    = (const float*)d_in[27];
  float* out = (float*)d_out;

  uintptr_t cur = (uintptr_t)d_ws;
  auto allocf = [&](size_t n) -> float* {
    float* r = (float*)cur; cur += (((n * 4) + 255) / 256) * 256; return r;
  };
  auto alloch = [&](size_t n) -> half_t* {
    half_t* r = (half_t*)cur; cur += (((n * 2) + 255) / 256) * 256; return r;
  };

  const int cCin[4]  = {1, 64, 128, 256};
  const int cCout[4] = {64, 128, 256, 512};
  const int cHin[4]  = {HIMG, 64, 32, 16};
  const int cWin[4]  = {WIMG, 256, 128, 64};
  const int cHout[4] = {64, 32, 16, 16};
  const int cWout[4] = {256, 128, 64, 64};
  const int cStr[4]  = {2, 2, 2, 1};
  int cKpad[4];
  for (int i = 0; i < 4; ++i) cKpad[i] = ((cCin[i] * 9 + 31) / 32) * 32;

  half_t* pcw[4];
  for (int i = 0; i < 4; ++i) pcw[i] = alloch((size_t)cKpad[i] * cCout[i]);
  half_t* pWih_enc = alloch((size_t)HID * 4 * HID);
  half_t* pWhh_enc = alloch((size_t)HID * 4 * HID);
  half_t* pWenc    = alloch((size_t)HID * HID);
  half_t* pWdec    = alloch((size_t)HID * HID);
  half_t* pWih0    = alloch((size_t)(EMBd + HID) * 4 * HID);
  half_t* pWhh0    = alloch((size_t)HID * 4 * HID);
  half_t* pWih1    = alloch((size_t)HID * 4 * HID);
  half_t* pWhh1    = alloch((size_t)HID * 4 * HID);
  half_t* pOutW    = alloch((size_t)HID * VOCAB);

  float*  actA   = allocf((size_t)Bn * 64 * 64 * 256);
  float*  actB   = allocf((size_t)Bn * 512 * 16 * 64);
  half_t* enc_x  = alloch((size_t)Wp * Nenc * HID);
  float*  enc_h  = allocf((size_t)Nenc * HID);
  float*  enc_c  = allocf((size_t)Nenc * HID);
  half_t* enc_h16 = alloch((size_t)Nenc * HID);
  float*  gates_e = allocf((size_t)Nenc * 4 * HID);
  float*  enc_f32 = allocf((size_t)Bn * L * HID);
  half_t* enc_f16 = alloch((size_t)Bn * L * HID);
  float*  enc_proj = allocf((size_t)Bn * L * HID);
  float*  h0 = allocf((size_t)Bn * HID);
  float*  c0 = allocf((size_t)Bn * HID);
  float*  h1 = allocf((size_t)Bn * HID);
  float*  c1 = allocf((size_t)Bn * HID);
  half_t* h0_16 = alloch((size_t)Bn * HID);
  half_t* h1_16 = alloch((size_t)Bn * HID);
  float*  query  = allocf((size_t)Bn * HID);
  float*  scores = allocf((size_t)Bn * L);
  float*  attn   = allocf((size_t)Bn * L);
  float*  ctx    = allocf((size_t)Bn * HID);
  half_t* x0_16  = alloch((size_t)Bn * (EMBd + HID));
  float*  gates0 = allocf((size_t)Bn * 4 * HID);
  float*  gates1 = allocf((size_t)Bn * 4 * HID);
  float*  logits = allocf((size_t)Bn * VOCAB);
  float*  lossacc = allocf(1);

  // gemm: M%16==0, N%256==0, K%32==0
  auto gemm = [&](const half_t* A, int lda, const half_t* B,
                  float* C, int ldc, const float* bias, const float* bias2,
                  int M, int N, int K, int acc) {
    dim3 g(M / 16, N / 256);
    gemm_wmma<<<g, 128, 0, stream>>>(A, lda, B, C, ldc, bias, bias2, K, acc);
  };
  auto zf32 = [&](float* p, int n) {
    zero_f32<<<(n + 255) / 256, 256, 0, stream>>>(p, n);
  };
  auto zf16 = [&](half_t* p, int n) {
    zero_f16<<<(n + 255) / 256, 256, 0, stream>>>(p, n);
  };

  // --- pack weights ---
  for (int i = 0; i < 4; ++i) {
    int n = cKpad[i] * cCout[i];
    pack_conv_frag<<<(n + 255) / 256, 256, 0, stream>>>(cw[i], pcw[i],
                                                        cCout[i], cCin[i],
                                                        cKpad[i]);
  }
  auto packd = [&](const float* W, half_t* dst, int N, int K) {
    int n = N * K;
    pack_dense_frag<<<(n + 255) / 256, 256, 0, stream>>>(W, dst, N, K);
  };
  packd(enc_Wih, pWih_enc, 4 * HID, HID);
  packd(enc_Whh, pWhh_enc, 4 * HID, HID);
  packd(att_Wenc, pWenc, HID, HID);
  packd(att_Wdec, pWdec, HID, HID);
  packd(dec_Wih0, pWih0, 4 * HID, EMBd + HID);
  packd(dec_Whh0, pWhh0, 4 * HID, HID);
  packd(dec_Wih1, pWih1, 4 * HID, HID);
  packd(dec_Whh1, pWhh1, 4 * HID, HID);
  packd(out_W, pOutW, VOCAB, HID);

  // --- conv backbone (ping-pong; conv4 also emits enc_x f16) ---
  const float* cin_p[4]  = {images, actA, actB, actA};
  float*       cout_p[4] = {actA, actB, actA, actB};
  for (int i = 0; i < 4; ++i) {
    int Mtot = Bn * cHout[i] * cWout[i];
    dim3 g(Mtot / 16, cCout[i] / 64);
    conv_relu_wmma<<<g, 32, 0, stream>>>(
        cin_p[i], pcw[i], cb[i], cout_p[i],
        (i == 3) ? enc_x : (half_t*)nullptr,
        Bn, cCin[i], cHin[i], cWin[i],
        cCout[i], cHout[i], cWout[i], cStr[i], cKpad[i]);
  }

  // --- row encoder: 64-step LSTM ---
  zf32(enc_h, Nenc * HID);
  zf32(enc_c, Nenc * HID);
  zf16(enc_h16, Nenc * HID);
  for (int t = 0; t < Wp; ++t) {
    const half_t* xt = enc_x + (size_t)t * Nenc * HID;
    gemm(xt, HID, pWih_enc, gates_e, 4 * HID,
         enc_bih, enc_bhh, Nenc, 4 * HID, HID, 0);
    gemm(enc_h16, HID, pWhh_enc, gates_e, 4 * HID,
         nullptr, nullptr, Nenc, 4 * HID, HID, 1);
    int n = Nenc * HID;
    lstm_cell<<<(n + 255) / 256, 256, 0, stream>>>(
        gates_e, enc_h, enc_c, enc_h16, Nenc, HID,
        enc_f32, enc_f16, t, Hp, Wp, 1);
  }

  // --- enc_proj = enc @ att_Wenc^T ---
  gemm(enc_f16, HID, pWenc, enc_proj, HID, nullptr, nullptr,
       Bn * L, HID, HID, 0);

  // --- decoder: 128 teacher-forced steps ---
  zf32(h0, Bn * HID); zf32(c0, Bn * HID);
  zf32(h1, Bn * HID); zf32(c1, Bn * HID);
  zf16(h0_16, Bn * HID); zf16(h1_16, Bn * HID);
  zf32(lossacc, 1);

  for (int t = 0; t < TGT; ++t) {
    gemm(h1_16, HID, pWdec, query, HID, nullptr, nullptr,
         Bn, HID, HID, 0);
    attn_scores<<<dim3(L, Bn), 64, 0, stream>>>(enc_proj, query, att_v,
                                                scores, L, HID);
    softmax_rows<<<Bn, 256, 0, stream>>>(scores, attn, L);
    context_kernel<<<dim3(HID / 64, Bn), 64, 0, stream>>>(attn, enc_f32,
                                                          ctx, L, HID);
    {
      int n = Bn * (EMBd + HID);
      build_x0<<<(n + 255) / 256, 256, 0, stream>>>(targets, t, TGT, emb,
                                                    EMBd, ctx, x0_16, Bn, HID);
    }
    gemm(x0_16, EMBd + HID, pWih0, gates0, 4 * HID,
         dec_bih0, dec_bhh0, Bn, 4 * HID, EMBd + HID, 0);
    gemm(h0_16, HID, pWhh0, gates0, 4 * HID,
         nullptr, nullptr, Bn, 4 * HID, HID, 1);
    {
      int n = Bn * HID;
      lstm_cell<<<(n + 255) / 256, 256, 0, stream>>>(
          gates0, h0, c0, h0_16, Bn, HID, nullptr, nullptr, 0, 1, 1, 0);
    }
    gemm(h0_16, HID, pWih1, gates1, 4 * HID,
         dec_bih1, dec_bhh1, Bn, 4 * HID, HID, 0);
    gemm(h1_16, HID, pWhh1, gates1, 4 * HID,
         nullptr, nullptr, Bn, 4 * HID, HID, 1);
    {
      int n = Bn * HID;
      lstm_cell<<<(n + 255) / 256, 256, 0, stream>>>(
          gates1, h1, c1, h1_16, Bn, HID, nullptr, nullptr, 0, 1, 1, 0);
    }
    gemm(h1_16, HID, pOutW, logits, VOCAB, out_b, nullptr,
         Bn, VOCAB, HID, 0);
    nll_kernel<<<Bn, 256, 0, stream>>>(logits, targets, t, TGT, lossacc, VOCAB);
  }

  finalize_kernel<<<1, 1, 0, stream>>>(lossacc, out, 1.0f / (float)Bn);
}
